// DGCNN_32908039422448
// MI455X (gfx1250) — matrix operand order repaired and checked by reading, not compiled, not run on hardware
//
#include <hip/hip_runtime.h>
#include <hip/hip_bf16.h>

// ---------------------------------------------------------------------------
// DGCNN forward for MI455X (gfx1250, wave32, WMMA).
// Strategy:
//  * kNN Gram matrices via v_wmma_f32_16x16x32_f16 (f16 in, f32 accum),
//    fused with streaming top-k (per-lane register lists, LDS tile exchange,
//    s_wait_dscnt for wave-internal LDS ordering).
//  * EdgeConv decomposition: [xi, xj-xi]@W + b = xi@(Wa-Wb) + xj@Wb + b,
//    and max_j only touches the xj@Wb term -> two skinny WMMA GEMMs + gather-max.
// ---------------------------------------------------------------------------

typedef __attribute__((ext_vector_type(16))) _Float16 v16h;
typedef __attribute__((ext_vector_type(8)))  float    v8f;

#define NPTS 8192
#define FDIM 128
#define HDIM 256
#define CDIM 40
#define CPAD 48
#define KNN1 10
#define KNN2 8

#define BIGF 3.0e38f

// ---- WMMA fragment loaders (wave32 layouts per CDNA5 ISA 7.12.2) -----------
// A fragment 16x32 f16: lanes 0-15 row M=lane hold K {k0+0..7, k0+16..23},
// lanes 16-31 row M=lane-16 hold K {k0+8..15, k0+24..31}.
__device__ inline v16h load_frag_a(const _Float16* __restrict__ base, int ld,
                                   int row0, int k0) {
    int lane = threadIdx.x & 31;
    const _Float16* p = base + (size_t)(row0 + (lane & 15)) * ld + k0 + (lane >> 4) * 8;
    v16h f;
    uint4* fp = reinterpret_cast<uint4*>(&f);
    fp[0] = *reinterpret_cast<const uint4*>(p);        // K +0..7 (or +8..15)
    fp[1] = *reinterpret_cast<const uint4*>(p + 16);   // K +16..23 (or +24..31)
    return f;
}

// B fragment 32x16 f16 from a row-major [N][K] (i.e. B^T) matrix:
// lanes 0-15 col N=lane hold K {k0+0..7, k0+16..23}; lanes 16-31 hold K {k0+8..15, k0+24..31}.
__device__ inline v16h load_frag_b(const _Float16* __restrict__ baseT, int ld,
                                   int col0, int k0) {
    int lane = threadIdx.x & 31;
    const _Float16* p = baseT + (size_t)(col0 + (lane & 15)) * ld + k0 + (lane >> 4) * 8;
    v16h f;
    uint4* fp = reinterpret_cast<uint4*>(&f);
    fp[0] = *reinterpret_cast<const uint4*>(p);
    fp[1] = *reinterpret_cast<const uint4*>(p + 16);
    return f;
}

template <int K>
__device__ inline void topk_insert(float (&bd)[K], int (&bi)[K], float s, int j) {
    if (s < bd[K - 1]) {
        bd[K - 1] = s; bi[K - 1] = j;
#pragma unroll
        for (int u = K - 1; u > 0; --u) {
            if (bd[u] < bd[u - 1]) {
                float td = bd[u]; bd[u] = bd[u - 1]; bd[u - 1] = td;
                int   ti = bi[u]; bi[u] = bi[u - 1]; bi[u - 1] = ti;
            }
        }
    }
}

// ---- convert x -> f16, compute row squared norms ---------------------------
__global__ __launch_bounds__(128) void cvt_x_kernel(const float* __restrict__ x,
                                                    _Float16* __restrict__ xh,
                                                    float* __restrict__ xsq) {
    int i = blockIdx.x, k = threadIdx.x;
    float v = x[(size_t)i * FDIM + k];
    xh[(size_t)i * FDIM + k] = (_Float16)v;
    __shared__ float red[128];
    red[k] = v * v;
    __syncthreads();
    for (int s = 64; s > 0; s >>= 1) {
        if (k < s) red[k] += red[k + s];
        __syncthreads();
    }
    if (k == 0) xsq[i] = red[0];
}

// ---- weight prep: split, subtract, transpose to [N][K] f16 -----------------
__global__ void prep_w1_kernel(const float* __restrict__ W1,
                               _Float16* __restrict__ WAT, _Float16* __restrict__ WBT) {
    int t = blockIdx.x * blockDim.x + threadIdx.x;  // over HDIM*FDIM
    if (t >= HDIM * FDIM) return;
    int n = t / FDIM, k = t % FDIM;
    float top = W1[(size_t)k * HDIM + n];
    float bot = W1[(size_t)(k + FDIM) * HDIM + n];
    WAT[(size_t)n * FDIM + k] = (_Float16)(top - bot);
    WBT[(size_t)n * FDIM + k] = (_Float16)bot;
}

__global__ void prep_w2_kernel(const float* __restrict__ W2,
                               _Float16* __restrict__ WAT, _Float16* __restrict__ WBT) {
    int t = blockIdx.x * blockDim.x + threadIdx.x;  // over CPAD*HDIM
    if (t >= CPAD * HDIM) return;
    int n = t / HDIM, k = t % HDIM;
    float top = 0.f, bot = 0.f;
    if (n < CDIM) {
        top = W2[(size_t)k * CDIM + n];
        bot = W2[(size_t)(k + HDIM) * CDIM + n];
    }
    WAT[(size_t)n * HDIM + k] = (_Float16)(top - bot);
    WBT[(size_t)n * HDIM + k] = (_Float16)bot;
}

// ---- fused Gram(WMMA) + streaming top-k kNN --------------------------------
// Block = 128 threads (4 waves). Block owns 16 rows; each wave sweeps N/4
// columns in 16-wide tiles. Score = |xj|^2 - 2*dot (|xi|^2 constant per row).
template <int D, int K>
__global__ __launch_bounds__(128) void knn_kernel(const _Float16* __restrict__ X,
                                                  const float* __restrict__ xsq,
                                                  int* __restrict__ idx_out, int n) {
    constexpr int KF = D / 32;
    const int lane = threadIdx.x & 31;
    const int wave = threadIdx.x >> 5;
    const int r0   = blockIdx.x * 16;
    const int l15  = lane & 15;
    const int sel  = lane >> 4;

    __shared__ float tile[4][16][16];
    __shared__ float sd[4 * 32 * K];
    __shared__ int   si[4 * 32 * K];

    v16h afrag[KF];
#pragma unroll
    for (int f = 0; f < KF; ++f) afrag[f] = load_frag_a(X, D, r0, f * 32);

    float bestd[K]; int besti[K];
#pragma unroll
    for (int t = 0; t < K; ++t) { bestd[t] = BIGF; besti[t] = 0; }

    const int iglob = r0 + l15;         // row this lane scans in top-k phase
    const int tpw   = (n >> 4) >> 2;    // column tiles per wave

    for (int tt = 0; tt < tpw; ++tt) {
        int c0 = (wave * tpw + tt) * 16;
        v8f acc = {};
#pragma unroll
        for (int f = 0; f < KF; ++f) {
            v16h b = load_frag_b(X, D, c0, f * 32);
            acc = __builtin_amdgcn_wmma_f32_16x16x32_f16(
                false, afrag[f], false, b, (short)0, acc, false, false);
        }
        // C layout: lane col = l15; rows v+sel*8. Fold in |xj|^2, spill tile.
        float qj = xsq[c0 + l15];
#pragma unroll
        for (int v = 0; v < 8; ++v)
            tile[wave][v + sel * 8][l15] = qj - 2.0f * acc[v];
        asm volatile("s_wait_dscnt 0" ::: "memory");  // wave-internal LDS RAW
        // top-k scan: lane owns row l15, columns [sel*8, sel*8+8)
#pragma unroll
        for (int t = 0; t < 8; ++t) {
            int c = sel * 8 + t;
            float s = tile[wave][l15][c];
            int j = c0 + c;
            if (j != iglob) topk_insert<K>(bestd, besti, s, j);
        }
        asm volatile("s_wait_dscnt 0" ::: "memory");  // reads done before next spill
    }

    // merge: 4 waves x 2 half-lanes per row -> 8 lists of K
    int base = (wave * 32 + lane) * K;
#pragma unroll
    for (int t = 0; t < K; ++t) { sd[base + t] = bestd[t]; si[base + t] = besti[t]; }
    __syncthreads();
    if (threadIdx.x < 16) {
        int r = threadIdx.x;
        float md[K]; int mi[K];
#pragma unroll
        for (int t = 0; t < K; ++t) { md[t] = BIGF; mi[t] = 0; }
        for (int w = 0; w < 4; ++w)
            for (int h = 0; h < 2; ++h) {
                int b2 = (w * 32 + r + h * 16) * K;
                for (int t = 0; t < K; ++t)
                    topk_insert<K>(md, mi, sd[b2 + t], si[b2 + t]);
            }
#pragma unroll
        for (int t = 0; t < K; ++t) idx_out[(size_t)(r0 + r) * K + t] = mi[t];
    }
}

// ---- generic one-wave-per-tile WMMA GEMM: C[M][Nt] = A[M][K] * Bt[Nt][K]^T -
__global__ __launch_bounds__(32) void gemm_kernel(const _Float16* __restrict__ A,
                                                  const _Float16* __restrict__ Bt,
                                                  float* __restrict__ C,
                                                  int K, int Nt) {
    int m0 = blockIdx.x * 16;
    int n0 = blockIdx.y * 16;
    v8f acc = {};
    for (int k0 = 0; k0 < K; k0 += 32) {
        v16h a = load_frag_a(A, K, m0, k0);
        v16h b = load_frag_b(Bt, K, n0, k0);
        acc = __builtin_amdgcn_wmma_f32_16x16x32_f16(
            false, a, false, b, (short)0, acc, false, false);
    }
    int lane = threadIdx.x;
    int col  = n0 + (lane & 15);
    int sel  = lane >> 4;
#pragma unroll
    for (int v = 0; v < 8; ++v)
        C[(size_t)(m0 + v + sel * 8) * Nt + col] = acc[v];
}

// ---- EdgeConv1 epilogue: h = relu(A1 + b1 + max_j B1[idx]); emit f16 + |h|^2
__global__ __launch_bounds__(256) void edge1_kernel(const float* __restrict__ A1,
                                                    const float* __restrict__ B1,
                                                    const float* __restrict__ b1,
                                                    const int* __restrict__ idx,
                                                    _Float16* __restrict__ hh,
                                                    float* __restrict__ hsq) {
    int i = blockIdx.x, c = threadIdx.x;  // c in [0,256)
    float a = A1[(size_t)i * HDIM + c] + b1[c];
    float m = -BIGF;
#pragma unroll
    for (int j = 0; j < KNN1; ++j) {
        int jj = idx[(size_t)i * KNN1 + j];
        m = fmaxf(m, B1[(size_t)jj * HDIM + c]);
    }
    float v = fmaxf(a + m, 0.0f);
    hh[(size_t)i * HDIM + c] = (_Float16)v;
    __shared__ float red[256];
    red[c] = v * v;
    __syncthreads();
    for (int s = 128; s > 0; s >>= 1) {
        if (c < s) red[c] += red[c + s];
        __syncthreads();
    }
    if (c == 0) hsq[i] = red[0];
}

// ---- EdgeConv2 epilogue: out = A2 + b2 + max_j B2[idx] ---------------------
__global__ __launch_bounds__(64) void edge2_kernel(const float* __restrict__ A2,
                                                   const float* __restrict__ B2,
                                                   const float* __restrict__ b2,
                                                   const int* __restrict__ idx,
                                                   float* __restrict__ out) {
    int i = blockIdx.x, c = threadIdx.x;
    if (c >= CDIM) return;
    float a = A2[(size_t)i * CPAD + c] + b2[c];
    float m = -BIGF;
#pragma unroll
    for (int j = 0; j < KNN2; ++j) {
        int jj = idx[(size_t)i * KNN2 + j];
        m = fmaxf(m, B2[(size_t)jj * CPAD + c]);
    }
    out[(size_t)i * CDIM + c] = a + m;
}

extern "C" void kernel_launch(void* const* d_in, const int* in_sizes, int n_in,
                              void* d_out, int out_size, void* d_ws, size_t ws_size,
                              hipStream_t stream) {
    const float* x  = (const float*)d_in[0];
    // d_in[1] = edge_index (unused by the reference computation)
    const float* W1 = (const float*)d_in[2];
    const float* b1 = (const float*)d_in[3];
    const float* W2 = (const float*)d_in[4];
    const float* b2 = (const float*)d_in[5];
    float* out = (float*)d_out;

    char* p = (char*)d_ws;
    auto carve = [&](size_t bytes) {
        void* r = (void*)p;
        p += (bytes + 255) & ~(size_t)255;
        return r;
    };
    _Float16* xh   = (_Float16*)carve((size_t)NPTS * FDIM * 2);
    float*    xsq  = (float*)   carve((size_t)NPTS * 4);
    int*      idx1 = (int*)     carve((size_t)NPTS * KNN1 * 4);
    _Float16* W1AT = (_Float16*)carve((size_t)HDIM * FDIM * 2);
    _Float16* W1BT = (_Float16*)carve((size_t)HDIM * FDIM * 2);
    float*    A1   = (float*)   carve((size_t)NPTS * HDIM * 4);
    float*    B1   = (float*)   carve((size_t)NPTS * HDIM * 4);
    _Float16* hh   = (_Float16*)carve((size_t)NPTS * HDIM * 2);
    float*    hsq  = (float*)   carve((size_t)NPTS * 4);
    int*      idx2 = (int*)     carve((size_t)NPTS * KNN2 * 4);
    _Float16* W2AT = (_Float16*)carve((size_t)CPAD * HDIM * 2);
    _Float16* W2BT = (_Float16*)carve((size_t)CPAD * HDIM * 2);
    float*    A2   = (float*)   carve((size_t)NPTS * CPAD * 4);
    float*    B2   = (float*)   carve((size_t)NPTS * CPAD * 4);

    cvt_x_kernel<<<NPTS, 128, 0, stream>>>(x, xh, xsq);
    prep_w1_kernel<<<(HDIM * FDIM + 255) / 256, 256, 0, stream>>>(W1, W1AT, W1BT);
    prep_w2_kernel<<<(CPAD * HDIM + 255) / 256, 256, 0, stream>>>(W2, W2AT, W2BT);

    // layer 1
    knn_kernel<FDIM, KNN1><<<NPTS / 16, 128, 0, stream>>>(xh, xsq, idx1, NPTS);
    gemm_kernel<<<dim3(NPTS / 16, HDIM / 16), 32, 0, stream>>>(xh, W1AT, A1, FDIM, HDIM);
    gemm_kernel<<<dim3(NPTS / 16, HDIM / 16), 32, 0, stream>>>(xh, W1BT, B1, FDIM, HDIM);
    edge1_kernel<<<NPTS, 256, 0, stream>>>(A1, B1, b1, idx1, hh, hsq);

    // layer 2
    knn_kernel<HDIM, KNN2><<<NPTS / 16, 128, 0, stream>>>(hh, hsq, idx2, NPTS);
    gemm_kernel<<<dim3(NPTS / 16, CPAD / 16), 32, 0, stream>>>(hh, W2AT, A2, HDIM, CPAD);
    gemm_kernel<<<dim3(NPTS / 16, CPAD / 16), 32, 0, stream>>>(hh, W2BT, B2, HDIM, CPAD);
    edge2_kernel<<<NPTS, 64, 0, stream>>>(A2, B2, b2, idx2, out);
}
